// GATEncoder_798863917682
// MI455X (gfx1250) — matrix operand order repaired and checked by reading, not compiled, not run on hardware
//
#include <hip/hip_runtime.h>

// Problem constants (from reference setup_inputs)
#define B_   8     // batch graphs
#define NN   256   // nodes (fully connected)
#define FIN  128
#define H1_  4
#define C1_  128
#define D1_  512   // H1*C1
#define FOUT 128

typedef __attribute__((ext_vector_type(16))) __bf16       v16bf;
typedef __attribute__((ext_vector_type(8)))  float        v8f;
typedef __attribute__((ext_vector_type(4)))  unsigned int u32x4;

__device__ __forceinline__ unsigned short f2bf(float f) {
  // round-to-nearest-even fp32 -> bf16 (NaN edge ignored; inputs are finite)
  unsigned u = __float_as_uint(f);
  u += 0x7FFFu + ((u >> 16) & 1u);
  return (unsigned short)(u >> 16);
}

// ---------------- elementwise fp32 -> bf16 ----------------
__global__ void cvt_bf16_kernel(const float* __restrict__ in,
                                unsigned short* __restrict__ out, int n) {
  int t = blockIdx.x * blockDim.x + threadIdx.x;
  if (t < n) out[t] = f2bf(in[t]);
}

// ---------------- fp32 [R,C] -> bf16 [C,R] ----------------
__global__ void transpose_cvt_kernel(const float* __restrict__ in,
                                     unsigned short* __restrict__ out, int R, int C) {
  int t = blockIdx.x * blockDim.x + threadIdx.x;
  if (t >= R * C) return;
  int r = t / C, c = t - r * C;
  out[c * R + r] = f2bf(in[t]);
}

// --------- per-node attention logits: a_src/a_dst [g][n][h] ---------
__global__ void att_dots_kernel(const float* __restrict__ feat,   // [g][n][Hh*Cc] fp32
                                const float* __restrict__ att_s,  // [Hh*Cc]
                                const float* __restrict__ att_d,
                                float* __restrict__ a_src, float* __restrict__ a_dst,
                                int Hh, int Cc, int total) {
  int t = blockIdx.x * blockDim.x + threadIdx.x;
  if (t >= total) return;
  int h  = t % Hh;
  int gn = t / Hh;                      // g*NN + n
  const float* f  = feat + (size_t)gn * (Hh * Cc) + h * Cc;
  const float* as = att_s + h * Cc;
  const float* ad = att_d + h * Cc;
  float ss = 0.f, dd = 0.f;
  for (int c = 0; c < Cc; ++c) { float v = f[c]; ss += v * as[c]; dd += v * ad[c]; }
  a_src[t] = ss;
  a_dst[t] = dd;
}

// --------- dense-graph segment softmax == row softmax of e[d,s] ---------
// blockIdx.x = ((g*Hh + h)*NN + d), threadIdx.x = s. Writes alphaT[g][h][d][s] bf16.
__global__ void __launch_bounds__(NN)
softmax_rows_kernel(const float* __restrict__ a_src, const float* __restrict__ a_dst,
                    unsigned short* __restrict__ alphaT, int Hh) {
  int d  = blockIdx.x % NN;
  int gh = blockIdx.x / NN;
  int h  = gh % Hh;
  int g  = gh / Hh;
  int s  = threadIdx.x;
  float e = a_src[(g * NN + s) * Hh + h] + a_dst[(g * NN + d) * Hh + h];
  e = e >= 0.f ? e : 0.2f * e;           // leaky_relu, NEG_SLOPE=0.2
  __shared__ float red[NN];
  red[s] = e;
  __syncthreads();
  for (int off = NN / 2; off > 0; off >>= 1) {
    if (s < off) red[s] = fmaxf(red[s], red[s + off]);
    __syncthreads();
  }
  float m = red[0];
  __syncthreads();
  float p = __expf(e - m);
  red[s] = p;
  __syncthreads();
  for (int off = NN / 2; off > 0; off >>= 1) {
    if (s < off) red[s] += red[s + off];
    __syncthreads();
  }
  float inv = 1.0f / (red[0] + 1e-16f);
  alphaT[(size_t)blockIdx.x * NN + s] = f2bf(p * inv);
}

// ---------------- generic bf16 WMMA GEMM, 32x32 register-blocked ----------------
// C[M,Nc] = A[M,K] (row-major bf16) x BT[Nc,K] (B stored transposed, bf16).
// One wave per 32x32 output tile: 2 A-frags x 2 B-frags -> 4 accumulators,
// 4x v_wmma_f32_16x16x32_bf16 per K-step (A/B fragments each reused twice).
// Batch axis = (g,h) with independent element strides. Epilogue optionally:
// +bias, ReLU, fp32 row-major out, bf16 row-major out, bf16 transposed out.
union ABReg { v16bf v; u32x4 q[2]; };

__global__ void __launch_bounds__(128)
wmma_gemm_kernel(const unsigned short* __restrict__ A,
                 const unsigned short* __restrict__ BT,
                 int M, int Nc, int K, int Hh,
                 long aG, long aH, long bG, long bH,
                 float* outF, long ofG, long ofH, int ofRow,
                 unsigned short* outB, long obG, long obH, int obRow,
                 unsigned short* outT, long otG, long otH, int otRow,
                 const float* bias, int biasH, int relu, int totalTiles) {
  int wave = threadIdx.x >> 5;
  int lane = threadIdx.x & 31;
  int t = blockIdx.x * 4 + wave;
  if (t >= totalTiles) return;          // wave-uniform: EXEC stays all-ones
  int nT   = Nc >> 5;                   // 32-wide tiles
  int per  = (M >> 5) * nT;
  int bidx = t / per;
  int rt   = t - bidx * per;
  int tm = rt / nT, tn = rt - tm * nT;  // 32-unit tile coords
  int g = bidx / Hh, h = bidx - g * Hh;
  int half = lane >> 4, ml = lane & 15;

  // A 16x32 bf16 frag: lane m=ml, elems 0..7 -> K=half*8+0..7, elems 8..15 -> K=16+half*8+0..7
  const unsigned short* Arow0 =
      A + (size_t)g * aG + (size_t)h * aH + (size_t)(tm * 32 + ml) * K + half * 8;
  const unsigned short* Arow1 = Arow0 + (size_t)16 * K;
  // B 32x16 bf16 frag: lane n=ml, elems 0..7 -> K=half*8+0..7, elems 8..15 -> K=16+half*8+0..7 (BT is [N][K])
  const unsigned short* Brow0 =
      BT + (size_t)g * bG + (size_t)h * bH + (size_t)(tn * 32 + ml) * K + half * 8;
  const unsigned short* Brow1 = Brow0 + (size_t)16 * K;

  v8f acc00 = {}, acc01 = {}, acc10 = {}, acc11 = {};
  for (int k0 = 0; k0 < K; k0 += 32) {
    ABReg a0, a1, b0, b1;
    a0.q[0] = *(const u32x4*)(Arow0 + k0);
    a0.q[1] = *(const u32x4*)(Arow0 + k0 + 16);
    a1.q[0] = *(const u32x4*)(Arow1 + k0);
    a1.q[1] = *(const u32x4*)(Arow1 + k0 + 16);
    b0.q[0] = *(const u32x4*)(Brow0 + k0);
    b0.q[1] = *(const u32x4*)(Brow0 + k0 + 16);
    b1.q[0] = *(const u32x4*)(Brow1 + k0);
    b1.q[1] = *(const u32x4*)(Brow1 + k0 + 16);
    acc00 = __builtin_amdgcn_wmma_f32_16x16x32_bf16(false, a0.v, false, b0.v,
                                                    (short)0, acc00, false, false);
    acc01 = __builtin_amdgcn_wmma_f32_16x16x32_bf16(false, a0.v, false, b1.v,
                                                    (short)0, acc01, false, false);
    acc10 = __builtin_amdgcn_wmma_f32_16x16x32_bf16(false, a1.v, false, b0.v,
                                                    (short)0, acc10, false, false);
    acc11 = __builtin_amdgcn_wmma_f32_16x16x32_bf16(false, a1.v, false, b1.v,
                                                    (short)0, acc11, false, false);
  }

  // C/D layout: lane col = ml, VGPR r -> row r + 8*half
  int col0 = tn * 32 + ml;
  int col1 = col0 + 16;
  float bv0 = bias ? bias[h * biasH + col0] : 0.0f;
  float bv1 = bias ? bias[h * biasH + col1] : 0.0f;
#pragma unroll
  for (int mi = 0; mi < 2; ++mi) {
    v8f am0 = mi ? acc10 : acc00;
    v8f am1 = mi ? acc11 : acc01;
#pragma unroll
    for (int rr = 0; rr < 8; ++rr) {
      int row = tm * 32 + mi * 16 + rr + half * 8;
      float v0 = am0[rr] + bv0;
      float v1 = am1[rr] + bv1;
      if (relu) { v0 = fmaxf(v0, 0.0f); v1 = fmaxf(v1, 0.0f); }
      if (outF) {
        float* p = outF + (size_t)g * ofG + (size_t)h * ofH + (size_t)row * ofRow;
        p[col0] = v0; p[col1] = v1;
      }
      if (outB) {
        unsigned short* p = outB + (size_t)g * obG + (size_t)h * obH + (size_t)row * obRow;
        p[col0] = f2bf(v0); p[col1] = f2bf(v1);
      }
      if (outT) {
        unsigned short* p = outT + (size_t)g * otG + (size_t)h * otH + row;
        p[(size_t)col0 * otRow] = f2bf(v0); p[(size_t)col1 * otRow] = f2bf(v1);
      }
    }
  }
}

extern "C" void kernel_launch(void* const* d_in, const int* in_sizes, int n_in,
                              void* d_out, int out_size, void* d_ws, size_t ws_size,
                              hipStream_t stream) {
  const float* x   = (const float*)d_in[0];
  const float* W1  = (const float*)d_in[1];
  const float* as1 = (const float*)d_in[2];
  const float* ad1 = (const float*)d_in[3];
  const float* b1  = (const float*)d_in[4];
  const float* W2  = (const float*)d_in[5];
  const float* as2 = (const float*)d_in[6];
  const float* ad2 = (const float*)d_in[7];
  const float* b2  = (const float*)d_in[8];
  float* out = (float*)d_out;
  (void)in_sizes; (void)n_in; (void)out_size; (void)ws_size;

  // workspace carve-out (~16 MB total, fits easily in the 192 MB L2)
  char* ws = (char*)d_ws;
  size_t off = 0;
  auto alloc = [&](size_t bytes) -> char* {
    char* p = ws + off;
    off = (off + bytes + 255) & ~(size_t)255;
    return p;
  };
  unsigned short* xb    = (unsigned short*)alloc((size_t)B_ * NN * FIN * 2);
  unsigned short* W1t   = (unsigned short*)alloc((size_t)FIN * D1_ * 2);     // [D1][FIN]
  unsigned short* W2t   = (unsigned short*)alloc((size_t)D1_ * FOUT * 2);    // [FOUT][D1]
  float*          h1f   = (float*)alloc((size_t)B_ * NN * D1_ * 4);          // [g][n][D1]
  unsigned short* h1T   = (unsigned short*)alloc((size_t)B_ * D1_ * NN * 2); // [g][feat][n]
  float*          asrc1 = (float*)alloc((size_t)B_ * NN * H1_ * 4);
  float*          adst1 = (float*)alloc((size_t)B_ * NN * H1_ * 4);
  unsigned short* alT1  = (unsigned short*)alloc((size_t)B_ * H1_ * NN * NN * 2); // [g][h][d][s]
  unsigned short* h1b   = (unsigned short*)alloc((size_t)B_ * NN * D1_ * 2);      // [g][n][D1]
  float*          h2f   = (float*)alloc((size_t)B_ * NN * FOUT * 4);
  unsigned short* h2T   = (unsigned short*)alloc((size_t)B_ * FOUT * NN * 2);
  float*          asrc2 = (float*)alloc((size_t)B_ * NN * 4);
  float*          adst2 = (float*)alloc((size_t)B_ * NN * 4);
  unsigned short* alT2  = (unsigned short*)alloc((size_t)B_ * NN * NN * 2);

  // 1) x -> bf16
  { int n = B_ * NN * FIN;
    cvt_bf16_kernel<<<(n + 255) / 256, 256, 0, stream>>>(x, xb, n); }
  // 2-3) weights -> transposed bf16 (K-contiguous B operands)
  transpose_cvt_kernel<<<(FIN * D1_ + 255) / 256, 256, 0, stream>>>(W1, W1t, FIN, D1_);
  transpose_cvt_kernel<<<(D1_ * FOUT + 255) / 256, 256, 0, stream>>>(W2, W2t, D1_, FOUT);

  // 4) GEMM1: h = x @ W1  -> h1f fp32 + h1T bf16 (transposed for AGG1's B operand)
  { int tiles = B_ * (NN / 32) * (D1_ / 32);     // 1024
    wmma_gemm_kernel<<<tiles / 4, 128, 0, stream>>>(
        xb, W1t, NN, D1_, FIN, 1,
        (long)NN * FIN, 0, 0, 0,
        h1f, (long)NN * D1_, 0, D1_,
        nullptr, 0, 0, 0,
        h1T, (long)D1_ * NN, 0, NN,
        nullptr, 0, 0, tiles); }

  // 5) attention logits layer 1
  { int tot = B_ * NN * H1_;
    att_dots_kernel<<<(tot + 255) / 256, 256, 0, stream>>>(h1f, as1, ad1, asrc1, adst1, H1_, C1_, tot); }
  // 6) row softmax -> alphaT1 bf16
  softmax_rows_kernel<<<B_ * H1_ * NN, NN, 0, stream>>>(asrc1, adst1, alT1, H1_);

  // 7) AGG1: h1 = relu(alphaT1 @ h + b1) -> h1b bf16 row-major (A of GEMM2)
  { int tiles = B_ * H1_ * (NN / 32) * (C1_ / 32);  // 1024
    wmma_gemm_kernel<<<tiles / 4, 128, 0, stream>>>(
        alT1, h1T, NN, C1_, NN, H1_,
        (long)H1_ * NN * NN, (long)NN * NN, (long)D1_ * NN, (long)C1_ * NN,
        nullptr, 0, 0, 0,
        h1b, (long)NN * D1_, C1_, D1_,
        nullptr, 0, 0, 0,
        b1, C1_, 1, tiles); }

  // 8) GEMM2: h2 = h1 @ W2 -> h2f fp32 + h2T bf16
  { int tiles = B_ * (NN / 32) * (FOUT / 32);    // 256
    wmma_gemm_kernel<<<tiles / 4, 128, 0, stream>>>(
        h1b, W2t, NN, FOUT, D1_, 1,
        (long)NN * D1_, 0, 0, 0,
        h2f, (long)NN * FOUT, 0, FOUT,
        nullptr, 0, 0, 0,
        h2T, (long)FOUT * NN, 0, NN,
        nullptr, 0, 0, tiles); }

  // 9) attention logits layer 2
  { int tot = B_ * NN;
    att_dots_kernel<<<(tot + 255) / 256, 256, 0, stream>>>(h2f, as2, ad2, asrc2, adst2, 1, FOUT, tot); }
  // 10) row softmax -> alphaT2 bf16
  softmax_rows_kernel<<<B_ * NN, NN, 0, stream>>>(asrc2, adst2, alT2, 1);

  // 11) AGG2: out = alphaT2 @ h2 + b2 -> d_out fp32
  { int tiles = B_ * (NN / 32) * (FOUT / 32);    // 256
    wmma_gemm_kernel<<<tiles / 4, 128, 0, stream>>>(
        alT2, h2T, NN, FOUT, NN, 1,
        (long)NN * NN, 0, (long)FOUT * NN, 0,
        out, (long)NN * FOUT, 0, FOUT,
        nullptr, 0, 0, 0,
        nullptr, 0, 0, 0,
        b2, 0, 0, tiles); }
}
